// DHCN_72129680769639
// MI455X (gfx1250) — matrix operand order repaired and checked by reading, not compiled, not run on hardware
//
#include <hip/hip_runtime.h>
#include <hip/hip_bf16.h>

typedef __attribute__((ext_vector_type(16))) _Float16 v16h;
typedef __attribute__((ext_vector_type(8)))  float    v8f;

#define N_NODES 100000
#define KADJ    8
#define DIM     128
#define BATCH   512
#define SEQ     50

// ---------- WMMA fragment packing (ISA 7.12.2 layouts, wave32) ----------
// A operand 16x32 f16: lane = hh*16 + M, vgpr v<4 -> K0 = 8*hh + 2v,
// v>=4 -> K0 = 16 + 8*hh + 2(v-4); halfs h[2v],h[2v+1] = K0,K0+1.
// Inverse map from (m,k) to packed index (fragments stored lane-contiguous):
__device__ __forceinline__ int apack_idx(int m, int k) {
    const int mt = m >> 4, lm = m & 15;
    const int kt = k >> 5, kk = k & 31;
    const int parity = kk & 1, k0 = kk & 30;
    const int hh = (k0 >> 3) & 1;                       // bit3 of k0
    const int v  = ((k0 & 16) >> 2) + ((k0 & 7) >> 1);  // +4 if k0>=16
    const int lane = hh * 16 + lm;
    return ((mt * (BATCH / 32) + kt) * 32 + lane) * 16 + 2 * v + parity;
}
// B operand 32x16 f16: lane = hh*16 + N, half r<8 holds K = 8*hh + r, r>=8 holds K = 16 + 8*hh + (r-8).
__device__ __forceinline__ int bpack_idx(int k, int n) {
    const int kt = k >> 5, kk = k & 31;
    const int nt = n >> 4, ln = n & 15;
    const int hh = (kk >> 3) & 1, r = (kk & 7) | ((kk >> 1) & 8);
    const int lane = hh * 16 + ln;
    return ((kt * (DIM / 16) + nt) * 32 + lane) * 16 + r;
}

// ---------------- hypergraph channel (bandwidth bound, float4) ----------------

// 4 nodes per 128-thread block; one wave per node, float4 per lane (512B/row/wave)
__global__ void spmm_step(const float* __restrict__ in,
                          const int* __restrict__ adj_idx,
                          const float* __restrict__ adj_val,
                          float* __restrict__ out) {
    const int n    = blockIdx.x * 4 + (threadIdx.x >> 5);
    const int lane = threadIdx.x & 31;
    const int*   ir = adj_idx + n * KADJ;
    const float* vr = adj_val + n * KADJ;
    float4 acc = make_float4(0.f, 0.f, 0.f, 0.f);
#pragma unroll
    for (int k = 0; k < KADJ; ++k) {
        const float4 x = ((const float4*)(in + (long)ir[k] * DIM))[lane];
        const float  w = vr[k];
        acc.x += w * x.x; acc.y += w * x.y; acc.z += w * x.z; acc.w += w * x.w;
    }
    ((float4*)(out + (long)n * DIM))[lane] = acc;
}

// hop 2 fused with (emb + cur1 + cur2)/3 -> h_global
__global__ void spmm_step_combine(const float* __restrict__ cur1,
                                  const float* __restrict__ emb,
                                  const int* __restrict__ adj_idx,
                                  const float* __restrict__ adj_val,
                                  float* __restrict__ hglobal) {
    const int n    = blockIdx.x * 4 + (threadIdx.x >> 5);
    const int lane = threadIdx.x & 31;
    const int*   ir = adj_idx + n * KADJ;
    const float* vr = adj_val + n * KADJ;
    float4 acc = make_float4(0.f, 0.f, 0.f, 0.f);
#pragma unroll
    for (int k = 0; k < KADJ; ++k) {
        const float4 x = ((const float4*)(cur1 + (long)ir[k] * DIM))[lane];
        const float  w = vr[k];
        acc.x += w * x.x; acc.y += w * x.y; acc.z += w * x.z; acc.w += w * x.w;
    }
    const float4 e = ((const float4*)(emb  + (long)n * DIM))[lane];
    const float4 c = ((const float4*)(cur1 + (long)n * DIM))[lane];
    float4 r;
    r.x = (e.x + c.x + acc.x) * (1.f / 3.f);
    r.y = (e.y + c.y + acc.y) * (1.f / 3.f);
    r.z = (e.z + c.z + acc.z) * (1.f / 3.f);
    r.w = (e.w + c.w + acc.w) * (1.f / 3.f);
    ((float4*)(hglobal + (long)n * DIM))[lane] = r;
}

// dst[r,:] = src[idx[r],:]  (4 rows per block, float4 per lane)
__global__ void gather_rows(const float* __restrict__ src,
                            const int* __restrict__ idx,
                            float* __restrict__ dst) {
    const int r    = blockIdx.x * 4 + (threadIdx.x >> 5);
    const int lane = threadIdx.x & 31;
    ((float4*)(dst + (long)r * DIM))[lane] =
        ((const float4*)(src + (long)idx[r] * DIM))[lane];
}

// ---------------- line-graph channel ----------------

// s[b,d] = masked mean of emb[items[b,l],d]; also emits packed f16 B-fragments of s
__global__ void session_mean(const float* __restrict__ emb,
                             const int* __restrict__ items,
                             const unsigned char* __restrict__ mask,
                             float* __restrict__ s,
                             _Float16* __restrict__ s_pack) {
    const int b    = blockIdx.x;
    const int lane = threadIdx.x;         // 32 lanes, 4 cols each
    float4 acc = make_float4(0.f, 0.f, 0.f, 0.f);
    float  den = 0.f;
    for (int l = 0; l < SEQ; ++l) {
        const unsigned char m = mask[b * SEQ + l];
        den += (float)m;
        if (m) {
            const float4 x = ((const float4*)(emb + (long)items[b * SEQ + l] * DIM))[lane];
            acc.x += x.x; acc.y += x.y; acc.z += x.z; acc.w += x.w;
        }
    }
    const float inv = 1.f / den;
    const float v0 = acc.x * inv, v1 = acc.y * inv, v2 = acc.z * inv, v3 = acc.w * inv;
    ((float4*)(s + b * DIM))[lane] = make_float4(v0, v1, v2, v3);
    const int d = lane * 4;
    s_pack[bpack_idx(b, d + 0)] = (_Float16)v0;
    s_pack[bpack_idx(b, d + 1)] = (_Float16)v1;
    s_pack[bpack_idx(b, d + 2)] = (_Float16)v2;
    s_pack[bpack_idx(b, d + 3)] = (_Float16)v3;
}

// one block per row i; thread j computes overlap(i,j), row-normalizes,
// emits DA directly as packed f16 A-fragments
__global__ void build_da_f16(const int* __restrict__ seq,
                             const unsigned char* __restrict__ mask,
                             _Float16* __restrict__ da_pack) {
    __shared__ int   sA[SEQ];
    __shared__ int   mA[SEQ];
    __shared__ float red[BATCH];
    const int i = blockIdx.x;
    const int j = threadIdx.x;
    if (j < SEQ) { sA[j] = seq[i * SEQ + j]; mA[j] = (int)mask[i * SEQ + j]; }
    __syncthreads();

    float ov = 0.f, msj = 0.f;
    for (int l = 0; l < SEQ; ++l) {
        const int aj = seq[j * SEQ + l];
        const int mj = (int)mask[j * SEQ + l];
        msj += (float)mj;
        ov  += (float)((sA[l] == aj) & (mA[l] != 0) & (mj != 0));
    }
    float a = ov / (2.f * msj - ov);   // A[i,j] = ov / (2*msum[j] - ov)
    if (i == j) a += 1.f;

    red[j] = a;
    __syncthreads();
    for (int off = BATCH / 2; off > 0; off >>= 1) {
        if (j < off) red[j] += red[j + off];
        __syncthreads();
    }
    da_pack[apack_idx(i, j)] = (_Float16)(a / red[0]);
}

// D[512,128] = A[512,512] * B[512,128] from packed f16 fragments.
// One wave per 16x16 tile; inner loop = two 32B fragment loads + one WMMA.
// Optionally re-emits D as packed B-fragments for the next GEMM.
__global__ void gemm_wmma_f16(const _Float16* __restrict__ apack,
                              const _Float16* __restrict__ bpack,
                              float* __restrict__ D,
                              _Float16* __restrict__ packD) {
    const int mt   = blockIdx.x;   // 0..31
    const int nt   = blockIdx.y;   // 0..7
    const int lane = threadIdx.x;  // 0..31
    const v16h* af = (const v16h*)apack;
    const v16h* bf = (const v16h*)bpack;

    v8f c = {};
#pragma unroll 4
    for (int kt = 0; kt < BATCH / 32; ++kt) {
        const v16h fa = af[(mt * (BATCH / 32) + kt) * 32 + lane];
        const v16h fb = bf[(kt * (DIM / 16) + nt) * 32 + lane];
        c = __builtin_amdgcn_wmma_f32_16x16x32_f16(false, fa, false, fb,
                                                   (short)0, c, false, false);
    }

    const int hh = lane >> 4, lm = lane & 15;
#pragma unroll
    for (int r = 0; r < 8; ++r) {
        const int row = mt * 16 + hh * 8 + r;   // C/D layout: vgpr r -> M = 8*hh + r
        const int col = nt * 16 + lm;
        D[row * DIM + col] = c[r];
        if (packD) packD[bpack_idx(row, col)] = (_Float16)c[r];
    }
}

__global__ void combine3(const float* __restrict__ a,
                         const float* __restrict__ b,
                         const float* __restrict__ c,
                         float* __restrict__ out, int n4) {
    const int i = blockIdx.x * blockDim.x + threadIdx.x;
    if (i < n4) {
        const float4 x = ((const float4*)a)[i];
        const float4 y = ((const float4*)b)[i];
        const float4 z = ((const float4*)c)[i];
        float4 r;
        r.x = (x.x + y.x + z.x) * (1.f / 3.f);
        r.y = (x.y + y.y + z.y) * (1.f / 3.f);
        r.z = (x.z + y.z + z.z) * (1.f / 3.f);
        r.w = (x.w + y.w + z.w) * (1.f / 3.f);
        ((float4*)out)[i] = r;
    }
}

// ---------------- driver ----------------

extern "C" void kernel_launch(void* const* d_in, const int* in_sizes, int n_in,
                              void* d_out, int out_size, void* d_ws, size_t ws_size,
                              hipStream_t stream) {
    const int*           items   = (const int*)d_in[0];
    const int*           seq     = (const int*)d_in[1];
    const unsigned char* mask    = (const unsigned char*)d_in[2];
    const int*           adj_idx = (const int*)d_in[3];
    const float*         adj_val = (const float*)d_in[4];
    const float*         emb     = (const float*)d_in[5];

    float* out      = (float*)d_out;
    float* out_hlg  = out;                                   // [512,128]
    float* out_hloc = out + BATCH * DIM;                     // [512,50,128]
    float* out_hg   = out + BATCH * DIM + BATCH * SEQ * DIM; // [100000,128]

    char* ws = (char*)d_ws;
    float*    s         = (float*)(ws + 0);          // 512*128*4
    float*    cur1      = (float*)(ws + 262144);     // 512*128*4
    float*    cur2      = (float*)(ws + 524288);     // 512*128*4
    _Float16* s_pack    = (_Float16*)(ws + 786432);  // 512*128*2
    _Float16* cur1_pack = (_Float16*)(ws + 917504);  // 512*128*2
    _Float16* da_pack   = (_Float16*)(ws + 1048576); // 512*512*2
    float*    g1        = (float*)(ws + 1572864);    // 100000*128*4 = 51.2MB

    // hypergraph conv (2 hops; L2-resident float4 gathers) + row gathers
    spmm_step<<<N_NODES / 4, 128, 0, stream>>>(emb, adj_idx, adj_val, g1);
    spmm_step_combine<<<N_NODES / 4, 128, 0, stream>>>(g1, emb, adj_idx, adj_val, out_hg);
    gather_rows<<<(BATCH * SEQ) / 4, 128, 0, stream>>>(out_hg, items, out_hloc);

    // line-graph conv: s (+packed), DA (packed), two WMMA GEMMs, combine
    session_mean<<<BATCH, 32, 0, stream>>>(emb, items, mask, s, s_pack);
    build_da_f16<<<BATCH, BATCH, 0, stream>>>(seq, mask, da_pack);
    gemm_wmma_f16<<<dim3(BATCH / 16, DIM / 16), 32, 0, stream>>>(da_pack, s_pack, cur1, cur1_pack);
    gemm_wmma_f16<<<dim3(BATCH / 16, DIM / 16), 32, 0, stream>>>(da_pack, cur1_pack, cur2, (_Float16*)nullptr);
    combine3<<<(BATCH * DIM / 4 + 255) / 256, 256, 0, stream>>>(s, cur1, cur2, out_hlg, BATCH * DIM / 4);
}
